// Attention_59579786330220
// MI455X (gfx1250) — matrix operand order repaired and checked by reading, not compiled, not run on hardware
//
#include <hip/hip_runtime.h>
#include <hip/hip_bf16.h>
#include <math.h>

// ---- problem constants ----
#define BB 8
#define NN 1024
#define CC 768
#define HH 12
#define HD 64
#define C3 (3*CC)          // 2304
#define MROWS (BB*NN)      // 8192
#define EPSV 1e-6f

typedef __attribute__((ext_vector_type(16))) _Float16 v16h;
typedef __attribute__((ext_vector_type(8)))  _Float16 v8h;
typedef __attribute__((ext_vector_type(8)))  float    v8f;
typedef __attribute__((ext_vector_type(4)))  unsigned int u32x4;
typedef __attribute__((ext_vector_type(8)))  int i32x8;
typedef __attribute__((ext_vector_type(4)))  int i32x4;

__device__ __forceinline__ v16h cat8(v8h lo, v8h hi) {
    return __builtin_shufflevector(lo, hi, 0,1,2,3,4,5,6,7,8,9,10,11,12,13,14,15);
}
__device__ __forceinline__ v8f wmma_f16(v16h a, v16h b, v8f c) {
    return __builtin_amdgcn_wmma_f32_16x16x32_f16(false, a, false, b, (short)0, c, false, false);
}

// ------------------------------------------------------------------
// TDM: DMA a 2D tile (nrows x row_elems of 2-byte elems, row stride
// row_stride_elems) from global memory into LDS, row-major, contiguous.
// D# layout per CDNA5 ISA ch.8.
// ------------------------------------------------------------------
__device__ __forceinline__ void tdm_load_2d_to_lds(unsigned int lds_off, unsigned long long gaddr,
                                                   unsigned int row_elems, unsigned int nrows,
                                                   unsigned int row_stride_elems) {
    u32x4 g0;
    g0.x = 1u;                                              // count=1, user descriptor
    g0.y = lds_off;                                         // lds_addr (bytes)
    g0.z = (unsigned int)(gaddr & 0xffffffffu);             // global_addr[31:0]
    g0.w = (unsigned int)((gaddr >> 32) & 0x1ffffffu) | (2u << 30); // addr[56:32] | type=2
    i32x8 g1;
    g1[0] = (int)(1u << 16);                                // wg_mask=0, data_size=1 (2B)
    g1[1] = (int)((row_elems & 0xffffu) << 16);             // tensor_dim0[15:0]
    g1[2] = (int)(((row_elems >> 16) & 0xffffu) | ((nrows & 0xffffu) << 16)); // dim0 hi | dim1 lo
    g1[3] = (int)(((nrows >> 16) & 0xffffu) | ((row_elems & 0xffffu) << 16)); // dim1 hi | tile_dim0
    g1[4] = (int)(nrows & 0xffffu);                         // tile_dim1 | tile_dim2=0
    g1[5] = (int)row_stride_elems;                          // tensor_dim0_stride[31:0]
    g1[6] = 0;
    g1[7] = 0;
    i32x4 z4 = {0, 0, 0, 0};
#if defined(__clang_major__) && (__clang_major__ >= 23)
    i32x8 z8 = {0, 0, 0, 0, 0, 0, 0, 0};
    __builtin_amdgcn_tensor_load_to_lds(g0, g1, z4, z4, z8, 0);
#else
    __builtin_amdgcn_tensor_load_to_lds(g0, g1, z4, z4, 0);
#endif
}

// ------------------------------------------------------------------
// Kernel 0: fp32 -> fp16 convert (grid-stride)
// ------------------------------------------------------------------
__global__ void cvt_f32_f16(const float* __restrict__ src, _Float16* __restrict__ dst, int n) {
    int i = blockIdx.x * blockDim.x + threadIdx.x;
    int stride = gridDim.x * blockDim.x;
    for (; i < n; i += stride) dst[i] = (_Float16)src[i];
}

// ------------------------------------------------------------------
// Kernel 1: QKV = Xh(8192x768) @ Wqkvh(2304x768)^T -> f16 (8192x2304)
// Ping-pong software pipeline: two register sets alternate across
// even/odd k-steps, so loads stay one step ahead of the WMMAs with
// zero rotation copies.
// ------------------------------------------------------------------
__global__ void __launch_bounds__(256)
qkv_gemm(const _Float16* __restrict__ Xh, const _Float16* __restrict__ Wh,
         _Float16* __restrict__ QKVh) {
    const int lane = threadIdx.x & 31, wave = threadIdx.x >> 5;
    const int col = lane & 15, hi = lane >> 4;
    const int tile = blockIdx.x * 8 + wave;          // 0 .. 18431
    const int MT = MROWS / 16;                        // 512
    const int mt = tile % MT, nt = tile / MT;         // nt < 36
    const int m0 = mt * 16, n0 = nt * 64;

    const _Float16* arow = Xh + (size_t)(m0 + col) * CC;
    const _Float16* brow = Wh + (size_t)(n0 + col) * CC + 16 * hi;

    auto ldA = [&](int k0) {
        const _Float16* pa = arow + k0 + 16 * hi;
        return cat8(*(const v8h*)pa, *(const v8h*)(pa + 8));
    };
    auto ldB = [&](int j, int k0) {
        return *(const v16h*)(brow + (size_t)j * 16 * CC + k0);
    };

    v8f acc[4] = {};
    v16h aA = ldA(0);
    v16h bA[4] = { ldB(0, 0), ldB(1, 0), ldB(2, 0), ldB(3, 0) };
    for (int k0 = 0; k0 < CC; k0 += 64) {            // two k-steps per iter
        const int k1 = k0 + 32;
        v16h aB = ldA(k1);
        v16h bB[4] = { ldB(0, k1), ldB(1, k1), ldB(2, k1), ldB(3, k1) };
#pragma unroll
        for (int j = 0; j < 4; ++j) acc[j] = wmma_f16(aA, bA[j], acc[j]);
        const int k2 = (k0 + 64 < CC) ? k0 + 64 : k0;   // clamped tail prefetch
        aA = ldA(k2);
#pragma unroll
        for (int j = 0; j < 4; ++j) bA[j] = ldB(j, k2);
#pragma unroll
        for (int j = 0; j < 4; ++j) acc[j] = wmma_f16(aB, bB[j], acc[j]);
    }
#pragma unroll
    for (int j = 0; j < 4; ++j) {
        const int n = n0 + j * 16 + col;
        const float sc = (n < CC) ? 0.125f : 1.0f;   // scale q by hd^-0.5
#pragma unroll
        for (int r = 0; r < 8; ++r) {
            const int m = m0 + r + 8 * hi;
            QKVh[(size_t)m * C3 + n] = (_Float16)(acc[j][r] * sc);
        }
    }
}

// ------------------------------------------------------------------
// Kernel 2: vsum[b,h,d] = sum_n V[b,h,n,d]   (for the eps/N term)
// ------------------------------------------------------------------
__global__ void vsum_kernel(const _Float16* __restrict__ QKVh, float* __restrict__ vsum) {
    int idx = blockIdx.x * blockDim.x + threadIdx.x;
    if (idx >= BB * HH * HD) return;
    const int d = idx & 63;
    const int h = (idx >> 6) % HH;
    const int b = idx / (HD * HH);
    const _Float16* p = QKVh + (size_t)(b * NN) * C3 + 2 * CC + h * HD + d;
    float s = 0.f;
    for (int n = 0; n < NN; ++n) s += (float)p[(size_t)n * C3];
    vsum[idx] = s;
}

// ------------------------------------------------------------------
// Kernel 3: flash-style masked attention.
// Block = 256 thr (8 waves) per (b, h, 128-query block); wave = 16 queries.
// Per 32-key block: K tile DMA'd to LDS by the Tensor Data Mover
// (wave 0: tensor_load_to_lds + s_wait_tensorcnt), V staged transposed
// in LDS, S via WMMA, online softmax via shfl_xor, O += P@V via WMMA.
// ------------------------------------------------------------------
__global__ void __launch_bounds__(256)
attn_kernel(const _Float16* __restrict__ QKVh, const float* __restrict__ policy,
            const float* __restrict__ vsum, _Float16* __restrict__ Oh) {
    __shared__ __attribute__((aligned(64))) _Float16 Kl[32 * HD];    // (key,d) row-major, TDM dest
    __shared__ __attribute__((aligned(64))) _Float16 Vt[HD * 32];    // (d,key) transposed
    __shared__ __attribute__((aligned(64))) _Float16 Pl[8][16 * 32]; // per-wave P tile

    const int lane = threadIdx.x & 31, wave = threadIdx.x >> 5;
    const int col = lane & 15, hi = lane >> 4;
    const int blk  = blockIdx.x;              // B*H*8
    const int qblk = blk % 8;
    const int h    = (blk / 8) % HH;
    const int b    = blk / (8 * HH);
    const int qbase = qblk * 128 + wave * 16;

    const _Float16* Qbase = QKVh + (size_t)(b * NN) * C3 + h * HD;
    const _Float16* Kbase = Qbase + CC;
    const _Float16* Vbase = Qbase + 2 * CC;

    // persistent Q A-fragments (hd=64 -> two K=32 chunks)
    v16h qa[2];
    {
        const _Float16* qrow = Qbase + (size_t)(qbase + col) * C3 + 16 * hi;
#pragma unroll
        for (int c = 0; c < 2; ++c)
            qa[c] = cat8(*(const v8h*)(qrow + c * 32), *(const v8h*)(qrow + c * 32 + 8));
    }
    float pq[8], mrun[8], srun[8];
#pragma unroll
    for (int r = 0; r < 8; ++r) {
        pq[r]   = policy[b * NN + qbase + r + 8 * hi];
        mrun[r] = -INFINITY;
        srun[r] = 0.f;
    }
    v8f oacc[4] = {};

    for (int kb = 0; kb < NN; kb += 32) {
        __syncthreads();                       // prior-iteration Kl/Vt reads done
        if (wave == 0) {
            // DMA K tile: 32 rows x 64 halfs, row stride 2304 halfs -> LDS row-major
            tdm_load_2d_to_lds((unsigned int)(uintptr_t)(void*)Kl,
                               (unsigned long long)(uintptr_t)(Kbase + (size_t)kb * C3),
                               HD, 32, C3);
        }
        // stage V transposed (threads), overlapping the TDM
        for (int t = threadIdx.x; t < 32 * HD; t += 256) {
            const int key = t >> 6, d = t & 63;
            Vt[d * 32 + key] = Vbase[(size_t)(kb + key) * C3 + d];
        }
        // prefetch next key-block's K/V rows while we compute this one
        if (kb + 32 < NN) {
            __builtin_prefetch(Kbase + (size_t)(kb + 32 + lane) * C3, 0, 1);
            __builtin_prefetch(Vbase + (size_t)(kb + 32 + lane) * C3, 0, 1);
        }
        if (wave == 0) __builtin_amdgcn_s_wait_tensorcnt(0);
        __syncthreads();

        // ---- S = Q K^T for two 16-key subtiles (K B-fragments from LDS) ----
        float ssub[2][8];
#pragma unroll
        for (int sub = 0; sub < 2; ++sub) {
            v8f sacc = {};
            const _Float16* krow = &Kl[(sub * 16 + col) * HD + 16 * hi];
#pragma unroll
            for (int c = 0; c < 2; ++c) {
                v16h kf = *(const v16h*)(krow + c * 32);
                sacc = wmma_f16(qa[c], kf, sacc);
            }
#pragma unroll
            for (int r = 0; r < 8; ++r) ssub[sub][r] = sacc[r];
        }

        const float pk0 = policy[b * NN + kb + col];
        const float pk1 = policy[b * NN + kb + 16 + col];

        // ---- online softmax per row ----
#pragma unroll
        for (int r = 0; r < 8; ++r) {
            float lm = fmaxf(ssub[0][r], ssub[1][r]);
#pragma unroll
            for (int off = 1; off < 16; off <<= 1) lm = fmaxf(lm, __shfl_xor(lm, off, 32));
            const float mn  = fmaxf(mrun[r], lm);
            const float fac = __expf(mrun[r] - mn);
            mrun[r] = mn;

            const int qg  = qbase + r + 8 * hi;
            const int kg0 = kb + col, kg1 = kb + 16 + col;
            float e0 = __expf(ssub[0][r] - mn) * (pq[r] * pk0 + ((kg0 == qg) ? (1.f - pk0) : 0.f));
            float e1 = __expf(ssub[1][r] - mn) * (pq[r] * pk1 + ((kg1 == qg) ? (1.f - pk1) : 0.f));
            float rs = e0 + e1;
#pragma unroll
            for (int off = 1; off < 16; off <<= 1) rs += __shfl_xor(rs, off, 32);
            srun[r] = srun[r] * fac + rs;
#pragma unroll
            for (int j = 0; j < 4; ++j) oacc[j][r] *= fac;

            Pl[wave][(r + 8 * hi) * 32 + col]      = (_Float16)e0;
            Pl[wave][(r + 8 * hi) * 32 + 16 + col] = (_Float16)e1;
        }

        // ---- O += P @ V  (P via LDS re-layout; V transposed in LDS) ----
        const _Float16* prow = &Pl[wave][col * 32 + 16 * hi];
        v16h pf = cat8(*(const v8h*)prow, *(const v8h*)(prow + 8));
#pragma unroll
        for (int j = 0; j < 4; ++j) {
            const _Float16* vrow = &Vt[(j * 16 + col) * 32 + 16 * hi];
            v16h vf = *(const v16h*)vrow;
            oacc[j] = wmma_f16(pf, vf, oacc[j]);
        }
    }

    // ---- finalize: (acc + (eps/N)*sum_v) / (s + eps) ----
    const float epsN = EPSV / (float)NN;
#pragma unroll
    for (int j = 0; j < 4; ++j) {
        const int d = j * 16 + col;
        const float vs = vsum[(b * HH + h) * HD + d];
#pragma unroll
        for (int r = 0; r < 8; ++r) {
            const int qg = qbase + r + 8 * hi;
            const float val = (oacc[j][r] + epsN * vs) / (srun[r] + EPSV);
            Oh[(size_t)(b * NN + qg) * CC + h * HD + d] = (_Float16)val;
        }
    }
}

// ------------------------------------------------------------------
// Kernel 4: Out = Oh(8192x768) @ Wprojh(768x768)^T + bproj -> f32
// Same ping-pong pipeline as qkv_gemm.
// ------------------------------------------------------------------
__global__ void __launch_bounds__(256)
proj_gemm(const _Float16* __restrict__ Oh, const _Float16* __restrict__ Wh,
          const float* __restrict__ bias, float* __restrict__ Out) {
    const int lane = threadIdx.x & 31, wave = threadIdx.x >> 5;
    const int col = lane & 15, hi = lane >> 4;
    const int tile = blockIdx.x * 8 + wave;          // 0 .. 6143
    const int MT = MROWS / 16;                        // 512
    const int mt = tile % MT, nt = tile / MT;         // nt < 12
    const int m0 = mt * 16, n0 = nt * 64;

    const _Float16* arow = Oh + (size_t)(m0 + col) * CC;
    const _Float16* brow = Wh + (size_t)(n0 + col) * CC + 16 * hi;

    auto ldA = [&](int k0) {
        const _Float16* pa = arow + k0 + 16 * hi;
        return cat8(*(const v8h*)pa, *(const v8h*)(pa + 8));
    };
    auto ldB = [&](int j, int k0) {
        return *(const v16h*)(brow + (size_t)j * 16 * CC + k0);
    };

    v8f acc[4] = {};
    v16h aA = ldA(0);
    v16h bA[4] = { ldB(0, 0), ldB(1, 0), ldB(2, 0), ldB(3, 0) };
    for (int k0 = 0; k0 < CC; k0 += 64) {
        const int k1 = k0 + 32;
        v16h aB = ldA(k1);
        v16h bB[4] = { ldB(0, k1), ldB(1, k1), ldB(2, k1), ldB(3, k1) };
#pragma unroll
        for (int j = 0; j < 4; ++j) acc[j] = wmma_f16(aA, bA[j], acc[j]);
        const int k2 = (k0 + 64 < CC) ? k0 + 64 : k0;
        aA = ldA(k2);
#pragma unroll
        for (int j = 0; j < 4; ++j) bA[j] = ldB(j, k2);
#pragma unroll
        for (int j = 0; j < 4; ++j) acc[j] = wmma_f16(aB, bB[j], acc[j]);
    }
#pragma unroll
    for (int j = 0; j < 4; ++j) {
        const int n = n0 + j * 16 + col;
        const float bj = bias[n];
#pragma unroll
        for (int r = 0; r < 8; ++r) {
            const int m = m0 + r + 8 * hi;
            Out[(size_t)m * CC + n] = acc[j][r] + bj;
        }
    }
}

// ------------------------------------------------------------------
extern "C" void kernel_launch(void* const* d_in, const int* in_sizes, int n_in,
                              void* d_out, int out_size, void* d_ws, size_t ws_size,
                              hipStream_t stream) {
    const float* x      = (const float*)d_in[0];   // (B,N,C)
    const float* policy = (const float*)d_in[1];   // (B,N,1)
    const float* Wqkv   = (const float*)d_in[2];   // (3C,C)
    const float* Wproj  = (const float*)d_in[3];   // (C,C)
    const float* bproj  = (const float*)d_in[4];   // (C,)
    float* out = (float*)d_out;

    // workspace carve-up (all 256B aligned)
    char* ws = (char*)d_ws;
    size_t off = 0;
    auto carve = [&](size_t bytes) { char* p = ws + off; off += (bytes + 255) & ~(size_t)255; return p; };
    _Float16* Xh     = (_Float16*)carve((size_t)MROWS * CC * 2);   // 12.6 MB
    _Float16* Wqkvh  = (_Float16*)carve((size_t)C3 * CC * 2);      //  3.5 MB
    _Float16* Wprojh = (_Float16*)carve((size_t)CC * CC * 2);      //  1.2 MB
    _Float16* QKVh   = (_Float16*)carve((size_t)MROWS * C3 * 2);   // 37.7 MB
    _Float16* Oh     = (_Float16*)carve((size_t)MROWS * CC * 2);   // 12.6 MB
    float*    vsum   = (float*)   carve((size_t)BB * HH * HD * 4); // 24.6 KB

    // 0) converts
    cvt_f32_f16<<<512, 256, 0, stream>>>(x,     Xh,     MROWS * CC);
    cvt_f32_f16<<<512, 256, 0, stream>>>(Wqkv,  Wqkvh,  C3 * CC);
    cvt_f32_f16<<<256, 256, 0, stream>>>(Wproj, Wprojh, CC * CC);

    // 1) QKV GEMM: 512 * 36 wave-tiles / 8 waves = 2304 blocks
    qkv_gemm<<<2304, 256, 0, stream>>>(Xh, Wqkvh, QKVh);

    // 2) per-(b,h,d) V column sums
    vsum_kernel<<<(BB * HH * HD + 255) / 256, 256, 0, stream>>>(QKVh, vsum);

    // 3) attention: B*H*(N/128) = 768 blocks
    attn_kernel<<<BB * HH * 8, 256, 0, stream>>>(QKVh, policy, vsum, Oh);

    // 4) projection: 512 * 12 / 8 = 768 blocks
    proj_gemm<<<768, 256, 0, stream>>>(Oh, Wprojh, bproj, out);
}
